// TopoSchNet_48215302865485
// MI455X (gfx1250) — matrix operand order repaired and checked by reading, not compiled, not run on hardware
//
#include <hip/hip_runtime.h>
#include <hip/hip_bf16.h>

typedef __attribute__((ext_vector_type(16))) _Float16 v16h;
typedef __attribute__((ext_vector_type(8)))  _Float16 v8h;
typedef __attribute__((ext_vector_type(8)))  float    v8f;

#define HID 128
#define WAVES 4

// shifted softplus, numerically stable
__device__ __forceinline__ float ssp(float x) {
    return fmaxf(x, 0.0f) + log1pf(__expf(-fabsf(x))) - 0.69314718056f;
}

__device__ __forceinline__ v16h cat8(v8h lo, v8h hi) {
    return __builtin_shufflevector(lo, hi, 0,1,2,3,4,5,6,7,8,9,10,11,12,13,14,15);
}

// A-fragment (16x32 f16) from an LDS tile with 16B-aligned row stride (halves).
// ISA 7.12.2: lanes 0-15: M=lane, elems 0..7 = K0..7, elems 8..15 = K16..23
//             lanes16-31: M=lane-16, elems 0..7 = K8..15, elems 8..15 = K24..31
__device__ __forceinline__ v16h load_a_lds(const _Float16* base, int strideH, int kc, int lane) {
    int m  = lane & 15;
    int kb = kc * 32 + ((lane < 16) ? 0 : 8);
    const _Float16* row = base + m * strideH + kb;
    v8h lo = *(const v8h*)(row);
    v8h hi = *(const v8h*)(row + 16);
    return cat8(lo, hi);
}

// B-fragment from pre-swizzled global weight buffer: one 32B contiguous read per lane.
__device__ __forceinline__ v16h load_b_frag(const _Float16* wf, int kc, int nt, int lane) {
    return *(const v16h*)(wf + ((((kc * 8) + nt) * 32 + lane) << 4));
}

__device__ __forceinline__ v8f wmma16(v16h a, v16h b, v8f c) {
    return __builtin_amdgcn_wmma_f32_16x16x32_f16(false, a, false, b, (short)0, c, false, false);
}

// ---------------- weight swizzle: f32 [T,Kreal,128] -> f16 B-fragment layout ----------------
__global__ void k_castw(const float* __restrict__ src, _Float16* __restrict__ dst,
                        int Kreal, int KC) {
    int per = KC << 12;                       // KC*8*32*16
    int idx = blockIdx.x * blockDim.x + threadIdx.x;
    int t   = blockIdx.y;
    if (idx >= per) return;
    int i    = idx & 15;
    int lane = (idx >> 4) & 31;
    int nt   = (idx >> 9) & 7;
    int kc   = idx >> 12;
    int n = nt * 16 + (lane & 15);
    int k = kc * 32 + ((lane < 16) ? 0 : 8) + (i & 7) + ((i & 8) ? 16 : 0);
    float v = (k < Kreal) ? src[(size_t)t * Kreal * HID + (size_t)k * HID + n] : 0.0f;
    dst[(size_t)t * per + idx] = (_Float16)v;
}

// ---------------- h = emb[z] ----------------
__global__ void k_embed(const int* __restrict__ z, const float* __restrict__ emb,
                        float* __restrict__ h, int N) {
    int idx = blockIdx.x * blockDim.x + threadIdx.x;
    if (idx >= N * HID) return;
    int node = idx >> 7, c = idx & 127;
    h[idx] = emb[z[node] * HID + c];
}

// ---------------- xx = h @ lin1 (no bias); also zero agg ----------------
__global__ void __launch_bounds__(128) k_node_lin1(
        const float* __restrict__ h, const _Float16* __restrict__ wf,
        float* __restrict__ xx, float* __restrict__ agg, int N) {
    __shared__ alignas(16) _Float16 sA[WAVES][16 * 136];
    int lane = threadIdx.x & 31, w = threadIdx.x >> 5;
    int nbase = (blockIdx.x * WAVES + w) * 16;
    for (int idx = lane; idx < 16 * HID; idx += 32) {
        int m = idx >> 7, n = idx & 127;
        int node = nbase + m;
        float v = (node < N) ? h[(size_t)node * HID + n] : 0.0f;
        sA[w][m * 136 + n] = (_Float16)v;
    }
    __syncthreads();
    int nlo = lane & 15, hi8 = (lane >= 16) ? 8 : 0;
    v8f zer = {};
    v8f acc[8];
    #pragma unroll
    for (int nt = 0; nt < 8; ++nt) acc[nt] = zer;
    #pragma unroll
    for (int kc = 0; kc < 4; ++kc) {
        v16h a = load_a_lds(sA[w], 136, kc, lane);
        #pragma unroll
        for (int nt = 0; nt < 8; ++nt)
            acc[nt] = wmma16(a, load_b_frag(wf, kc, nt, lane), acc[nt]);
    }
    #pragma unroll
    for (int nt = 0; nt < 8; ++nt)
        #pragma unroll
        for (int r = 0; r < 8; ++r) {
            int m = r + hi8, node = nbase + m;
            if (node < N) {
                int n = nt * 16 + nlo;
                xx[(size_t)node * HID + n]  = acc[nt][r];
                agg[(size_t)node * HID + n] = 0.0f;
            }
        }
}

// ---------------- edge kernel: smearing -> MLP(50->128->128) -> *C *xx[src] -> atomicAdd agg[dst] ----------------
__global__ void __launch_bounds__(128) k_edge(
        const int* __restrict__ ei, const float* __restrict__ pos,
        const float* __restrict__ xx, float* __restrict__ agg,
        const _Float16* __restrict__ w1f, const _Float16* __restrict__ w2f,
        const float* __restrict__ b1, const float* __restrict__ b2, int E) {
    __shared__ alignas(16) _Float16 sE[WAVES][16 * 72];
    __shared__ alignas(16) _Float16 sI[WAVES][16 * 136];
    __shared__ float sC[WAVES][16];
    __shared__ int   sS[WAVES][16], sD[WAVES][16];
    int lane = threadIdx.x & 31, w = threadIdx.x >> 5;
    int ebase = (blockIdx.x * WAVES + w) * 16;

    __builtin_prefetch(w1f + (lane << 4), 0, 0);
    __builtin_prefetch(w2f + (lane << 4), 0, 0);

    // ---- stage 0: geometry + gaussian smearing into LDS ----
    int row = lane & 15;
    int e = ebase + row;
    bool valid = (e < E);
    int si = 0, di = 0;
    if (valid) { si = ei[e]; di = ei[E + e]; }
    float dx = pos[di * 3 + 0] - pos[si * 3 + 0];
    float dy = pos[di * 3 + 1] - pos[si * 3 + 1];
    float dz = pos[di * 3 + 2] - pos[si * 3 + 2];
    float d = sqrtf(dx * dx + dy * dy + dz * dz);
    if (lane < 16) {
        sC[w][row] = valid ? (0.5f * (__cosf(d * 0.6283185307f) + 1.0f)) : 0.0f;
        sS[w][row] = si;
        sD[w][row] = di;
    }
    int kb = (lane < 16) ? 0 : 32;
    #pragma unroll
    for (int k = 0; k < 32; ++k) {
        int kk = kb + k;
        float g = 0.0f;
        if (valid && kk < 50) {
            float x = d - (float)kk * (5.0f / 49.0f);
            g = __expf(-48.02f * x * x);   // -0.5/(5/49)^2
        }
        sE[w][row * 72 + kk] = (_Float16)g;
    }
    __syncthreads();

    int nlo = lane & 15, hi8 = (lane >= 16) ? 8 : 0;
    v8f acc[8];

    // ---- GEMM1: e_attr[16x64] @ W1[64x128], init with b1 ----
    #pragma unroll
    for (int nt = 0; nt < 8; ++nt) {
        float b = b1[nt * 16 + nlo];
        #pragma unroll
        for (int r = 0; r < 8; ++r) acc[nt][r] = b;
    }
    #pragma unroll
    for (int kc = 0; kc < 2; ++kc) {
        v16h a = load_a_lds(sE[w], 72, kc, lane);
        #pragma unroll
        for (int nt = 0; nt < 8; ++nt)
            acc[nt] = wmma16(a, load_b_frag(w1f, kc, nt, lane), acc[nt]);
    }
    // ssp -> f16 intermediate tile (LDS transpose to A layout)
    #pragma unroll
    for (int nt = 0; nt < 8; ++nt)
        #pragma unroll
        for (int r = 0; r < 8; ++r) {
            int m = r + hi8;
            sI[w][m * 136 + nt * 16 + nlo] = (_Float16)ssp(acc[nt][r]);
        }
    __syncthreads();

    // ---- GEMM2: inter[16x128] @ W2[128x128], init with b2 ----
    #pragma unroll
    for (int nt = 0; nt < 8; ++nt) {
        float b = b2[nt * 16 + nlo];
        #pragma unroll
        for (int r = 0; r < 8; ++r) acc[nt][r] = b;
    }
    #pragma unroll
    for (int kc = 0; kc < 4; ++kc) {
        v16h a = load_a_lds(sI[w], 136, kc, lane);
        #pragma unroll
        for (int nt = 0; nt < 8; ++nt)
            acc[nt] = wmma16(a, load_b_frag(w2f, kc, nt, lane), acc[nt]);
    }

    // ---- epilogue: Wf * C * xx[src] -> atomicAdd agg[dst] ----
    #pragma unroll
    for (int nt = 0; nt < 8; ++nt)
        #pragma unroll
        for (int r = 0; r < 8; ++r) {
            int m = r + hi8;
            int ee = ebase + m;
            if (ee < E) {
                int n = nt * 16 + nlo;
                float v = acc[nt][r] * sC[w][m] * xx[(size_t)sS[w][m] * HID + n];
                atomicAdd(&agg[(size_t)sD[w][m] * HID + n], v);
            }
        }
}

// ---------------- h += ssp(agg@lin2 + b2) @ int_lin + bl ----------------
__global__ void __launch_bounds__(128) k_node_update(
        float* __restrict__ h, const float* __restrict__ agg,
        const _Float16* __restrict__ w2f, const float* __restrict__ b2,
        const _Float16* __restrict__ wlf, const float* __restrict__ bl, int N) {
    __shared__ alignas(16) _Float16 sA[WAVES][16 * 136];
    __shared__ alignas(16) _Float16 sI[WAVES][16 * 136];
    int lane = threadIdx.x & 31, w = threadIdx.x >> 5;
    int nbase = (blockIdx.x * WAVES + w) * 16;
    for (int idx = lane; idx < 16 * HID; idx += 32) {
        int m = idx >> 7, n = idx & 127;
        int node = nbase + m;
        float v = (node < N) ? agg[(size_t)node * HID + n] : 0.0f;
        sA[w][m * 136 + n] = (_Float16)v;
    }
    __syncthreads();
    int nlo = lane & 15, hi8 = (lane >= 16) ? 8 : 0;
    v8f acc[8];
    #pragma unroll
    for (int nt = 0; nt < 8; ++nt) {
        float b = b2[nt * 16 + nlo];
        #pragma unroll
        for (int r = 0; r < 8; ++r) acc[nt][r] = b;
    }
    #pragma unroll
    for (int kc = 0; kc < 4; ++kc) {
        v16h a = load_a_lds(sA[w], 136, kc, lane);
        #pragma unroll
        for (int nt = 0; nt < 8; ++nt)
            acc[nt] = wmma16(a, load_b_frag(w2f, kc, nt, lane), acc[nt]);
    }
    #pragma unroll
    for (int nt = 0; nt < 8; ++nt)
        #pragma unroll
        for (int r = 0; r < 8; ++r) {
            int m = r + hi8;
            sI[w][m * 136 + nt * 16 + nlo] = (_Float16)ssp(acc[nt][r]);
        }
    __syncthreads();
    #pragma unroll
    for (int nt = 0; nt < 8; ++nt) {
        float b = bl[nt * 16 + nlo];
        #pragma unroll
        for (int r = 0; r < 8; ++r) acc[nt][r] = b;
    }
    #pragma unroll
    for (int kc = 0; kc < 4; ++kc) {
        v16h a = load_a_lds(sI[w], 136, kc, lane);
        #pragma unroll
        for (int nt = 0; nt < 8; ++nt)
            acc[nt] = wmma16(a, load_b_frag(wlf, kc, nt, lane), acc[nt]);
    }
    #pragma unroll
    for (int nt = 0; nt < 8; ++nt)
        #pragma unroll
        for (int r = 0; r < 8; ++r) {
            int m = r + hi8, node = nbase + m;
            if (node < N) {
                int n = nt * 16 + nlo;
                size_t ix = (size_t)node * HID + n;
                h[ix] = h[ix] + acc[nt][r];
            }
        }
}

// ---------------- zero geom ----------------
__global__ void k_zero(float* __restrict__ g, int G) {
    int i = blockIdx.x * blockDim.x + threadIdx.x;
    if (i < G) g[i] = 0.0f;
}

// ---------------- per-node readout MLP 128->64->1, segment-sum into geom ----------------
__global__ void __launch_bounds__(64) k_readout(
        const float* __restrict__ h, const float* __restrict__ w1, const float* __restrict__ b1,
        const float* __restrict__ w2, const float* __restrict__ b2,
        float* __restrict__ geom, int atoms) {
    __shared__ float red[64];
    int node = blockIdx.x;
    int j = threadIdx.x;
    const float* hr = h + (size_t)node * HID;
    float a = b1[j];
    #pragma unroll 8
    for (int k = 0; k < HID; ++k) a += hr[k] * w1[k * 64 + j];
    red[j] = ssp(a) * w2[j];
    __syncthreads();
    #pragma unroll
    for (int s = 32; s > 0; s >>= 1) {
        if (j < s) red[j] += red[j + s];
        __syncthreads();
    }
    if (j == 0) atomicAdd(&geom[node / atoms], red[0] + b2[0]);
}

// ---------------- topo MLP + final combine ----------------
__global__ void k_final(const float* __restrict__ tda, const float* __restrict__ geom,
                        const float* __restrict__ w1, const float* __restrict__ b1,
                        const float* __restrict__ w2, const float* __restrict__ b2,
                        const float* __restrict__ fw, const float* __restrict__ fb,
                        float* __restrict__ out, int G) {
    int g = blockIdx.x * blockDim.x + threadIdx.x;
    if (g >= G) return;
    float t0 = tda[g * 2 + 0], t1 = tda[g * 2 + 1];
    float hb[16];
    #pragma unroll
    for (int i = 0; i < 16; ++i) {
        float v = t0 * w1[i] + t1 * w1[16 + i] + b1[i];
        hb[i] = fmaxf(v, 0.0f);
    }
    float o = geom[g] * fw[0] + fb[0];
    #pragma unroll
    for (int j = 0; j < 16; ++j) {
        float s = b2[j];
        #pragma unroll
        for (int i = 0; i < 16; ++i) s += hb[i] * w2[i * 16 + j];
        o += s * fw[1 + j];
    }
    out[g] = o;
}

extern "C" void kernel_launch(void* const* d_in, const int* in_sizes, int n_in,
                              void* d_out, int out_size, void* d_ws, size_t ws_size,
                              hipStream_t stream) {
    const int*   z    = (const int*)d_in[0];
    const float* pos  = (const float*)d_in[1];
    const float* tda  = (const float*)d_in[3];
    const int*   ei   = (const int*)d_in[4];
    const float* emb  = (const float*)d_in[5];
    const float* w1   = (const float*)d_in[6];
    const float* b1   = (const float*)d_in[7];
    const float* w2   = (const float*)d_in[8];
    const float* b2   = (const float*)d_in[9];
    const float* lin1 = (const float*)d_in[10];
    const float* lin2 = (const float*)d_in[11];
    const float* l2b  = (const float*)d_in[12];
    const float* ilw  = (const float*)d_in[13];
    const float* ilb  = (const float*)d_in[14];
    const float* ow1  = (const float*)d_in[15];
    const float* ob1  = (const float*)d_in[16];
    const float* ow2  = (const float*)d_in[17];
    const float* ob2  = (const float*)d_in[18];
    const float* tw1  = (const float*)d_in[19];
    const float* tb1  = (const float*)d_in[20];
    const float* tw2  = (const float*)d_in[21];
    const float* tb2  = (const float*)d_in[22];
    const float* fw   = (const float*)d_in[23];
    const float* fb   = (const float*)d_in[24];

    int N = in_sizes[0];
    int G = in_sizes[3] / 2;
    int E = in_sizes[4] / 2;
    int atoms = N / G;

    char* p = (char*)d_ws;
    auto alloc = [&](size_t bytes) {
        char* r = p;
        p += (bytes + 255) & ~(size_t)255;
        return r;
    };
    float* h    = (float*)alloc((size_t)N * HID * 4);
    float* xx   = (float*)alloc((size_t)N * HID * 4);
    float* agg  = (float*)alloc((size_t)N * HID * 4);
    float* geom = (float*)alloc((size_t)G * 4);
    _Float16* w1f = (_Float16*)alloc((size_t)6 * 8192 * 2);
    _Float16* w2f = (_Float16*)alloc((size_t)6 * 16384 * 2);
    _Float16* l1f = (_Float16*)alloc((size_t)6 * 16384 * 2);
    _Float16* l2f = (_Float16*)alloc((size_t)6 * 16384 * 2);
    _Float16* ilf = (_Float16*)alloc((size_t)6 * 16384 * 2);

    // swizzle weights to f16 B-fragment layout (once per launch)
    k_castw<<<dim3((8192 + 255) / 256, 6), 256, 0, stream>>>(w1,  w1f, 50, 2);
    k_castw<<<dim3((16384 + 255) / 256, 6), 256, 0, stream>>>(w2,  w2f, 128, 4);
    k_castw<<<dim3((16384 + 255) / 256, 6), 256, 0, stream>>>(lin1, l1f, 128, 4);
    k_castw<<<dim3((16384 + 255) / 256, 6), 256, 0, stream>>>(lin2, l2f, 128, 4);
    k_castw<<<dim3((16384 + 255) / 256, 6), 256, 0, stream>>>(ilw,  ilf, 128, 4);

    k_embed<<<(N * HID + 255) / 256, 256, 0, stream>>>(z, emb, h, N);

    int nbN = (N + 16 * WAVES - 1) / (16 * WAVES);
    int nbE = (E + 16 * WAVES - 1) / (16 * WAVES);
    for (int t = 0; t < 6; ++t) {
        k_node_lin1<<<nbN, 128, 0, stream>>>(h, l1f + (size_t)t * 16384, xx, agg, N);
        k_edge<<<nbE, 128, 0, stream>>>(ei, pos, xx, agg,
                                        w1f + (size_t)t * 8192, w2f + (size_t)t * 16384,
                                        b1 + t * HID, b2 + t * HID, E);
        k_node_update<<<nbN, 128, 0, stream>>>(h, agg,
                                               l2f + (size_t)t * 16384, l2b + t * HID,
                                               ilf + (size_t)t * 16384, ilb + t * HID, N);
    }

    k_zero<<<(G + 255) / 256, 256, 0, stream>>>(geom, G);
    k_readout<<<N, 64, 0, stream>>>(h, ow1, ob1, ow2, ob2, geom, atoms);
    k_final<<<(G + 255) / 256, 256, 0, stream>>>(tda, geom, tw1, tb1, tw2, tb2, fw, fb,
                                                 (float*)d_out, G);
}
